// TM_HGNN_12214886989913
// MI455X (gfx1250) — matrix operand order repaired and checked by reading, not compiled, not run on hardware
//
#include <hip/hip_runtime.h>
#include <hip/hip_bf16.h>

// ---------------------------------------------------------------------------
// GCN (3x GCNConv + mean-pool + linear) for MI455X / gfx1250.
// GEMMs use v_wmma_f32_16x16x32_bf16 (wave32); aggregation is L2-resident
// float4 gathers + global fp32 atomics.
// ---------------------------------------------------------------------------

typedef __attribute__((ext_vector_type(16))) __bf16 v16bf;
typedef __attribute__((ext_vector_type(8)))  __bf16 v8bf;
typedef __attribute__((ext_vector_type(8)))  float  v8f;

#define HDIM 128

__device__ __forceinline__ unsigned short f2bf_rne(float x) {
    unsigned u = __float_as_uint(x);
    u += 0x7FFFu + ((u >> 16) & 1u);          // round-to-nearest-even
    return (unsigned short)(u >> 16);
}

// ---------------- elementwise helpers ----------------

__global__ void gcn_zero_f32(float* __restrict__ p, int n) {
    int i = blockIdx.x * 256 + threadIdx.x;
    if (i < n) p[i] = 0.0f;
}

// fp32 -> bf16 (4 elems/thread, row-major copy)
__global__ void gcn_cvt_bf16(const float* __restrict__ in,
                             unsigned short* __restrict__ out, int n4) {
    int i = blockIdx.x * 256 + threadIdx.x;
    if (i >= n4) return;
    float4 v = ((const float4*)in)[i];
    ushort4 r;
    r.x = f2bf_rne(v.x); r.y = f2bf_rne(v.y);
    r.z = f2bf_rne(v.z); r.w = f2bf_rne(v.w);
    ((ushort4*)out)[i] = r;
}

// Transpose+convert the three 128x128 weights: Wt[l][n][k] = bf16(W_l[k][n])
__global__ void gcn_cvt_weights(const float* __restrict__ W1,
                                const float* __restrict__ W2,
                                const float* __restrict__ W3,
                                unsigned short* __restrict__ Wt) {
    int idx = blockIdx.x * 256 + threadIdx.x;       // 3*128*128 = 49152
    if (idx >= 3 * HDIM * HDIM) return;
    int l = idx >> 14, r = idx & 16383;
    int n = r >> 7, k = r & 127;
    const float* W = (l == 0) ? W1 : (l == 1) ? W2 : W3;
    Wt[idx] = f2bf_rne(W[k * HDIM + n]);
}

// ---------------- degree / normalization ----------------

__global__ void gcn_degrees(const int* __restrict__ ei,    // [2,E]
                            const int* __restrict__ mask,  // [E]
                            float* __restrict__ deg,       // [3,N]
                            int nE, int nN) {
    int e = blockIdx.x * 256 + threadIdx.x;
    if (e >= nE) return;
    int d = ei[nE + e];
    atomicAdd(&deg[d], 1.0f);                       // layer 1: all edges
    int mk = mask[e];
    if (mk == 1)      atomicAdd(&deg[nN + d], 1.0f);
    else if (mk == 2) atomicAdd(&deg[2 * nN + d], 1.0f);
}

__global__ void gcn_deg_to_dinv(float* __restrict__ deg, int n3) {
    int i = blockIdx.x * 256 + threadIdx.x;
    if (i < n3) deg[i] = rsqrtf(deg[i] + 1.0f);     // self-loop adds 1
}

// ---------------- WMMA GEMM: Hlin[N,128] = Xb[N,128] @ Wt^T ----------------
// grid.x = N/16 row tiles, block = 256 (8 waves); wave w -> columns [16w,16w+16)

__global__ void gcn_gemm_wmma(const unsigned short* __restrict__ Xb,  // bf16 [N,128]
                              const unsigned short* __restrict__ Wtw, // bf16 [128,128] (n-major)
                              float* __restrict__ Hlin) {
    const int lane = threadIdx.x & 31;
    const int wave = threadIdx.x >> 5;              // column tile id (0..7)
    const int m    = lane & 15;
    const int half = lane >> 4;
    const long long row0 = (long long)blockIdx.x * 16;

    const __bf16* arow = (const __bf16*)Xb  + (row0 + m) * HDIM;
    const __bf16* bcol = (const __bf16*)Wtw + (long long)(wave * 16 + m) * HDIM;

    v8f c = {};
#pragma unroll
    for (int ks = 0; ks < 4; ++ks) {
        const int kbase = ks * 32;
        // A frag: lanes 0-15 hold K 0-15, lanes 16-31 hold K 16-31 (contiguous)
        v16bf a = *(const v16bf*)(arow + kbase + 16 * half);
        // B frag: lanes 0-15 hold K 0-15, lanes 16-31 hold K 16-31 (contiguous in Wt)
        v16bf b = *(const v16bf*)(bcol + kbase + 16 * half);
        c = __builtin_amdgcn_wmma_f32_16x16x32_bf16(
                false, a, false, b, (short)0, c, false, false);
    }
    // C layout: col = lane&15, VGPR i -> row 8*half + i
    float* cr = Hlin + row0 * HDIM + wave * 16 + m;
#pragma unroll
    for (int i = 0; i < 8; ++i)
        cr[(long long)(8 * half + i) * HDIM] = c[i];
}

// ---------------- aggregation ----------------

// agg[i,:] = hlin[i,:] * dinv[i]^2   (self-loop term; also initializes agg)
__global__ void gcn_agg_init(const float* __restrict__ hlin,
                             const float* __restrict__ dinv,
                             float* __restrict__ agg, int nN) {
    int i = blockIdx.x * 8 + (threadIdx.x >> 5);
    if (i >= nN) return;
    float di = dinv[i], cf = di * di;
    int lane = threadIdx.x & 31;
    float4 v = ((const float4*)(hlin + (long long)i * HDIM))[lane];
    float4 r = { v.x * cf, v.y * cf, v.z * cf, v.w * cf };
    ((float4*)(agg + (long long)i * HDIM))[lane] = r;
}

// per-edge: agg[dst,:] += hlin[src,:] * dinv[src]*dinv[dst]  (32 lanes/edge)
__global__ void gcn_edge_scatter(const int* __restrict__ ei,
                                 const int* __restrict__ mask,
                                 const float* __restrict__ dinv,
                                 const float* __restrict__ hlin,
                                 float* __restrict__ agg,
                                 int mode, int nE) {
    int e = blockIdx.x * 8 + (threadIdx.x >> 5);
    if (e >= nE) return;
    if (mode && mask[e] != mode) return;            // uniform per 32-lane group
    int s = ei[e], d = ei[nE + e];
    float cf = dinv[s] * dinv[d];
    int lane = threadIdx.x & 31;
    float4 v = ((const float4*)(hlin + (long long)s * HDIM))[lane];
    float* dp = agg + (long long)d * HDIM + lane * 4;
    atomicAdd(dp + 0, v.x * cf);
    atomicAdd(dp + 1, v.y * cf);
    atomicAdd(dp + 2, v.z * cf);
    atomicAdd(dp + 3, v.w * cf);
}

__global__ void gcn_bias_act(float* __restrict__ h, const float* __restrict__ b,
                             long long total, int do_relu) {
    long long idx = (long long)blockIdx.x * 256 + threadIdx.x;
    if (idx >= total) return;
    float v = h[idx] + b[idx & (HDIM - 1)];
    if (do_relu) v = v > 0.0f ? v : 0.0f;
    h[idx] = v;
}

// ---------------- pooling + head ----------------

__global__ void gcn_pool(const float* __restrict__ h, const int* __restrict__ batch,
                         float* __restrict__ sums, float* __restrict__ cnt, int nN) {
    int i = blockIdx.x * 8 + (threadIdx.x >> 5);
    if (i >= nN) return;
    int g = batch[i];
    int lane = threadIdx.x & 31;
    float4 v = ((const float4*)(h + (long long)i * HDIM))[lane];
    float* sp = sums + (long long)g * HDIM + lane * 4;
    atomicAdd(sp + 0, v.x); atomicAdd(sp + 1, v.y);
    atomicAdd(sp + 2, v.z); atomicAdd(sp + 3, v.w);
    if (lane == 0) atomicAdd(&cnt[g], 1.0f);
}

__global__ void gcn_head(const float* __restrict__ sums, const float* __restrict__ cnt,
                         const float* __restrict__ Wl, const float* __restrict__ bl,
                         float* __restrict__ out) {
    int g = blockIdx.x;
    int f = threadIdx.x;                            // 128 threads = 4 waves
    float c = cnt[g]; c = (c < 1.0f) ? 1.0f : c;
    float v = (sums[(long long)g * HDIM + f] / c) * Wl[f];
#pragma unroll
    for (int off = 16; off; off >>= 1) v += __shfl_down(v, off, 32);
    __shared__ float red[4];
    if ((f & 31) == 0) red[f >> 5] = v;
    __syncthreads();
    if (f == 0) out[g] = red[0] + red[1] + red[2] + red[3] + bl[0];
}

// ---------------------------------------------------------------------------

extern "C" void kernel_launch(void* const* d_in, const int* in_sizes, int n_in,
                              void* d_out, int out_size, void* d_ws, size_t ws_size,
                              hipStream_t stream) {
    const float* x    = (const float*)d_in[0];
    const int*   ei   = (const int*)  d_in[1];   // [2,E]
    const int*   mask = (const int*)  d_in[2];
    const int*   batc = (const int*)  d_in[3];
    const float* W1   = (const float*)d_in[4];
    const float* b1   = (const float*)d_in[5];
    const float* W2   = (const float*)d_in[6];
    const float* b2   = (const float*)d_in[7];
    const float* W3   = (const float*)d_in[8];
    const float* b3   = (const float*)d_in[9];
    const float* Wl   = (const float*)d_in[10];
    const float* bl   = (const float*)d_in[11];
    float* out = (float*)d_out;

    const int N = in_sizes[0] / HDIM;            // 50000
    const int E = in_sizes[2];                   // 600000
    const int G = out_size;                      // 64

    // ---- workspace carving (256B aligned) ----
    auto align256 = [](size_t v) { return (v + 255) & ~(size_t)255; };
    char* ws = (char*)d_ws;
    size_t off = 0;
    float* hA    = (float*)(ws + off); off = align256(off + (size_t)N * HDIM * 4);
    float* hlin  = (float*)(ws + off); off = align256(off + (size_t)N * HDIM * 4);
    unsigned short* xbf = (unsigned short*)(ws + off); off = align256(off + (size_t)N * HDIM * 2);
    float* deg   = (float*)(ws + off); off = align256(off + (size_t)3 * N * 4);   // becomes dinv
    unsigned short* Wt = (unsigned short*)(ws + off); off = align256(off + (size_t)3 * HDIM * HDIM * 2);
    float* sums  = (float*)(ws + off); off = align256(off + (size_t)G * HDIM * 4);
    float* cnt   = (float*)(ws + off); off = align256(off + (size_t)G * 4);
    (void)ws_size; (void)n_in;

    const long long NH = (long long)N * HDIM;

    // ---- normalization setup ----
    gcn_zero_f32<<<(3 * N + 255) / 256, 256, 0, stream>>>(deg, 3 * N);
    gcn_zero_f32<<<(G * HDIM + G + 255) / 256, 256, 0, stream>>>(sums, G * HDIM + G); // sums+cnt contiguous
    gcn_degrees<<<(E + 255) / 256, 256, 0, stream>>>(ei, mask, deg, E, N);
    gcn_deg_to_dinv<<<(3 * N + 255) / 256, 256, 0, stream>>>(deg, 3 * N);
    gcn_cvt_weights<<<(3 * HDIM * HDIM + 255) / 256, 256, 0, stream>>>(W1, W2, W3, Wt);

    const float* dinv1 = deg;
    const float* dinv2 = deg + N;
    const float* dinv3 = deg + 2 * N;

    const int n4 = (int)(NH / 4);
    const int gemmGrid = N / 16;                 // 3125, exact
    const int nodeGrid = (N + 7) / 8;
    const int edgeGrid = (E + 7) / 8;
    const int ewGrid   = (int)((NH + 255) / 256);

    // ---- layer 1 (all edges, ReLU) ----
    gcn_cvt_bf16<<<(n4 + 255) / 256, 256, 0, stream>>>(x, xbf, n4);
    gcn_gemm_wmma<<<gemmGrid, 256, 0, stream>>>(xbf, Wt, hlin);
    gcn_agg_init<<<nodeGrid, 256, 0, stream>>>(hlin, dinv1, hA, N);
    gcn_edge_scatter<<<edgeGrid, 256, 0, stream>>>(ei, mask, dinv1, hlin, hA, 0, E);
    gcn_bias_act<<<ewGrid, 256, 0, stream>>>(hA, b1, NH, 1);

    // ---- layer 2 (mask==1, ReLU) ----
    gcn_cvt_bf16<<<(n4 + 255) / 256, 256, 0, stream>>>(hA, xbf, n4);
    gcn_gemm_wmma<<<gemmGrid, 256, 0, stream>>>(xbf, Wt + HDIM * HDIM, hlin);
    gcn_agg_init<<<nodeGrid, 256, 0, stream>>>(hlin, dinv2, hA, N);
    gcn_edge_scatter<<<edgeGrid, 256, 0, stream>>>(ei, mask, dinv2, hlin, hA, 1, E);
    gcn_bias_act<<<ewGrid, 256, 0, stream>>>(hA, b2, NH, 1);

    // ---- layer 3 (mask==2, no ReLU) ----
    gcn_cvt_bf16<<<(n4 + 255) / 256, 256, 0, stream>>>(hA, xbf, n4);
    gcn_gemm_wmma<<<gemmGrid, 256, 0, stream>>>(xbf, Wt + 2 * HDIM * HDIM, hlin);
    gcn_agg_init<<<nodeGrid, 256, 0, stream>>>(hlin, dinv3, hA, N);
    gcn_edge_scatter<<<edgeGrid, 256, 0, stream>>>(ei, mask, dinv3, hlin, hA, 2, E);
    gcn_bias_act<<<ewGrid, 256, 0, stream>>>(hA, b3, NH, 0);

    // ---- mean pool + linear head ----
    gcn_pool<<<nodeGrid, 256, 0, stream>>>(hA, batc, sums, cnt, N);
    gcn_head<<<G, HDIM, 0, stream>>>(sums, cnt, Wl, bl, out);
}
